// MolGNN_23467701305419
// MI455X (gfx1250) — matrix operand order repaired and checked by reading, not compiled, not run on hardware
//
#include <hip/hip_runtime.h>
#include <hip/hip_bf16.h>

// MolGNN on MI455X (gfx1250, wave32). Heavy math via v_wmma_f32_16x16x32_f16
// (f16 inputs, f32 accumulation). f32 master data; f16 conversion during LDS
// staging. LDS layouts are fragment-contiguous so every WMMA operand is a pair
// of ds_load_b128. Staging is branchless (clamped addresses + selects) and
// load-batched (all global_load_b128 issued before any convert/ds_store so the
// loads stay in flight). All GEMMs here have K%4==0, N%4==0 when transB==0,
// lda/ldb%4==0.

typedef __attribute__((ext_vector_type(16))) _Float16 v16h;
typedef __attribute__((ext_vector_type(8)))  _Float16 v8h;
typedef __attribute__((ext_vector_type(4)))  _Float16 v4h;
typedef __attribute__((ext_vector_type(8)))  float    v8f;

union Frag16 { v16h v; v8h h[2]; };

#define TPB 256
#define BM 256          // block tile rows
#define BN 128          // block tile cols
#define BK 32           // K chunk
#define LDSK (BK + 8)   // LDS row stride in halves (40 -> 80B, 16B aligned)
// 8 waves arranged 4(M) x 2(N); each wave: 4x4 grid of 16x16 accumulators
// -> 16 WMMAs per K-step against 16 ds_load_b128 fragment loads.

// ---------------------------------------------------------------------------
// WMMA GEMM: C[M,N] = act( A[M,K] * op(B) + bias )
//   transB==0: B is [K,N] row-major ; transB==1: B is [N,K] row-major (A*B^T)
// ---------------------------------------------------------------------------
__launch_bounds__(TPB)
__global__ void wmma_gemm_kernel(const float* __restrict__ A,
                                 const float* __restrict__ B,
                                 const float* __restrict__ bias,
                                 float* __restrict__ C,
                                 int M, int N, int K,
                                 int lda, int ldb, int ldc,
                                 int transB, int doRelu)
{
    __shared__ _Float16 As[BM][LDSK];   // row-major: As[m][k]
    __shared__ _Float16 Bs[BN][LDSK];   // K-transposed: Bs[n][k]

    const int tid  = threadIdx.x;
    const int lane = tid & 31;
    const int wave = tid >> 5;            // 0..7
    const int wm   = wave >> 1;           // 0..3  (64-row group)
    const int wn   = wave & 1;            // 0..1  (64-col group)
    const int m0 = blockIdx.y * BM;
    const int n0 = blockIdx.x * BN;

    v8f acc[4][4];
#pragma unroll
    for (int a = 0; a < 4; ++a)
#pragma unroll
        for (int b = 0; b < 4; ++b)
#pragma unroll
            for (int i = 0; i < 8; ++i) acc[a][b][i] = 0.0f;

    const int kTiles = (K + BK - 1) / BK;
    for (int kt = 0; kt < kTiles; ++kt) {
        const int k0 = kt * BK;
        const int krem = K - k0;   // >= 4, multiple of 4

        // ---- Batch all global loads for this K-step (keep them in flight) ----
        float4 areg[8];
#pragma unroll
        for (int i = 0; i < 8; ++i) {
            int idx = tid + i * TPB;             // 0..2047 float4 slots
            int r  = idx >> 3;                   // row 0..255
            int c4 = (idx & 7) << 2;             // k 0,4,...,28
            int grc = ((m0 + r) < M) ? (m0 + r) : (M - 1);
            int c4c = ((c4 + 4) <= krem) ? c4 : 0;
            areg[i] = *reinterpret_cast<const float4*>(&A[(long)grc * lda + k0 + c4c]);
        }
        float4 breg[4];
        if (transB) {
#pragma unroll
            for (int i = 0; i < 4; ++i) {
                int idx = tid + i * TPB;         // 0..1023 float4 slots
                int c  = idx >> 3;               // col 0..127
                int k4 = (idx & 7) << 2;         // k 0,4,...,28
                int gnc = ((n0 + c) < N) ? (n0 + c) : (N - 1);
                int k4c = ((k4 + 4) <= krem) ? k4 : 0;
                breg[i] = *reinterpret_cast<const float4*>(&B[(long)gnc * ldb + k0 + k4c]);
            }
        } else {
#pragma unroll
            for (int i = 0; i < 4; ++i) {
                int idx = tid + i * TPB;         // 0..1023 float4 slots
                int r  = idx >> 5;               // k row 0..31
                int c4 = (idx & 31) << 2;        // n 0,4,...,124
                int gkc = (r < krem) ? (k0 + r) : (K - 1);
                int c4c = ((n0 + c4 + 4) <= N) ? c4 : 0;
                breg[i] = *reinterpret_cast<const float4*>(&B[(long)gkc * ldb + n0 + c4c]);
            }
        }

        // ---- Convert + store to LDS (validity masks recomputed, branchless) ----
#pragma unroll
        for (int i = 0; i < 8; ++i) {
            int idx = tid + i * TPB;
            int r  = idx >> 3;
            int c4 = (idx & 7) << 2;
            bool ok = ((m0 + r) < M) && ((c4 + 4) <= krem);
            v4h hv;
            hv[0] = ok ? (_Float16)areg[i].x : (_Float16)0.0f;
            hv[1] = ok ? (_Float16)areg[i].y : (_Float16)0.0f;
            hv[2] = ok ? (_Float16)areg[i].z : (_Float16)0.0f;
            hv[3] = ok ? (_Float16)areg[i].w : (_Float16)0.0f;
            *reinterpret_cast<v4h*>(&As[r][c4]) = hv;
        }
        if (transB) {
#pragma unroll
            for (int i = 0; i < 4; ++i) {
                int idx = tid + i * TPB;
                int c  = idx >> 3;
                int k4 = (idx & 7) << 2;
                bool ok = ((n0 + c) < N) && ((k4 + 4) <= krem);
                v4h hv;
                hv[0] = ok ? (_Float16)breg[i].x : (_Float16)0.0f;
                hv[1] = ok ? (_Float16)breg[i].y : (_Float16)0.0f;
                hv[2] = ok ? (_Float16)breg[i].z : (_Float16)0.0f;
                hv[3] = ok ? (_Float16)breg[i].w : (_Float16)0.0f;
                *reinterpret_cast<v4h*>(&Bs[c][k4]) = hv;
            }
        } else {
#pragma unroll
            for (int i = 0; i < 4; ++i) {
                int idx = tid + i * TPB;
                int r  = idx >> 5;
                int c4 = (idx & 31) << 2;
                bool ok = (r < krem) && ((n0 + c4 + 4) <= N);
                Bs[c4 + 0][r] = ok ? (_Float16)breg[i].x : (_Float16)0.0f;
                Bs[c4 + 1][r] = ok ? (_Float16)breg[i].y : (_Float16)0.0f;
                Bs[c4 + 2][r] = ok ? (_Float16)breg[i].z : (_Float16)0.0f;
                Bs[c4 + 3][r] = ok ? (_Float16)breg[i].w : (_Float16)0.0f;
            }
        }
        __syncthreads();

        // A fragments (ISA 7.12.2): lane row = lane&15, K-half by lane>>4;
        // each fragment = 2x ds_load_b128.
        const int kb  = (lane >> 4) << 3;   // 0 or 8
        const int kb2 = (lane >> 4) << 4;   // 0 or 16
        Frag16 afr[4];
#pragma unroll
        for (int mf = 0; mf < 4; ++mf) {
            const int ar = (wm << 6) + (mf << 4) + (lane & 15);
            afr[mf].h[0] = *reinterpret_cast<const v8h*>(&As[ar][kb]);
            afr[mf].h[1] = *reinterpret_cast<const v8h*>(&As[ar][16 + kb]);
        }
        // B fragments: same K order as A (kb, 16+kb); Bs is [n][k] -> contiguous.
#pragma unroll
        for (int nf = 0; nf < 4; ++nf) {
            Frag16 b;
            const int bn = (wn << 6) + (nf << 4) + (lane & 15);
            b.h[0] = *reinterpret_cast<const v8h*>(&Bs[bn][kb]);
            b.h[1] = *reinterpret_cast<const v8h*>(&Bs[bn][16 + kb]);
#pragma unroll
            for (int mf = 0; mf < 4; ++mf)
                acc[mf][nf] = __builtin_amdgcn_wmma_f32_16x16x32_f16(
                    false, afr[mf].v, false, b.v, (short)0, acc[mf][nf], false, false);
        }
        __syncthreads();
    }

    // Epilogue: C/D layout — VGPR r: lanes 0-15 -> M=r, lanes 16-31 -> M=8+r
#pragma unroll
    for (int nf = 0; nf < 4; ++nf) {
        const int col = n0 + (wn << 6) + (nf << 4) + (lane & 15);
        if (col >= N) continue;
        const float bv = bias ? bias[col] : 0.0f;
#pragma unroll
        for (int mf = 0; mf < 4; ++mf) {
            const int rowBase = m0 + (wm << 6) + (mf << 4) + ((lane >> 4) << 3);
#pragma unroll
            for (int r = 0; r < 8; ++r) {
                const int row = rowBase + r;
                if (row < M) {
                    float v = acc[mf][nf][r] + bv;
                    if (doRelu) v = fmaxf(v, 0.0f);
                    C[(long)row * ldc + col] = v;
                }
            }
        }
    }
}

// ---------------------------------------------------------------------------
// Supporting elementwise / irregular kernels
// ---------------------------------------------------------------------------
__global__ void fill_kernel(float* __restrict__ p, float v, long n) {
    long i = (long)blockIdx.x * TPB + threadIdx.x;
    if (i < n) p[i] = v;
}

// dst[n, Cout] = (c < Cin) ? maybe_relu(src[n, Cin]) : 0
__global__ void pad_relu_kernel(const float* __restrict__ src, float* __restrict__ dst,
                                long n, int Cin, int Cout, int doRelu) {
    long idx = (long)blockIdx.x * TPB + threadIdx.x;
    if (idx >= n * Cout) return;
    long node = idx / Cout; int c = (int)(idx % Cout);
    float v = 0.0f;
    if (c < Cin) {
        v = src[node * Cin + c];
        if (doRelu) v = fmaxf(v, 0.0f);
    }
    dst[idx] = v;
}

// agg[dst[e], :] += m[src[e], :]   (float4 granularity, HW fp32 atomics)
__global__ void scatter_add_kernel(const float* __restrict__ m, const int* __restrict__ ei,
                                   float* __restrict__ agg, int E, int C4) {
    long idx = (long)blockIdx.x * TPB + threadIdx.x;
    if (idx >= (long)E * C4) return;
    int e  = (int)(idx / C4);
    int c4 = (int)(idx % C4);
    int s = ei[e];        // edge_index row 0
    int d = ei[E + e];    // edge_index row 1
    float4 v = ((const float4*)m)[(long)s * C4 + c4];
    float* ap = agg + ((long)d * C4 + c4) * 4;
    unsafeAtomicAdd(ap + 0, v.x);
    unsafeAtomicAdd(ap + 1, v.y);
    unsafeAtomicAdd(ap + 2, v.z);
    unsafeAtomicAdd(ap + 3, v.w);
}

// GRUCell gates [r,z,n]; h updated in place
__global__ void gru_kernel(const float* __restrict__ gi, const float* __restrict__ gh,
                           float* __restrict__ h, long n, int C) {
    long idx = (long)blockIdx.x * TPB + threadIdx.x;
    if (idx >= n * C) return;
    long node = idx / C; int c = (int)(idx % C);
    const float* gir = gi + node * 3L * C;
    const float* ghr = gh + node * 3L * C;
    float ir = gir[c], iz = gir[C + c], in_ = gir[2 * C + c];
    float hr = ghr[c], hz = ghr[C + c], hn  = ghr[2 * C + c];
    float r  = 1.0f / (1.0f + __expf(-(ir + hr)));
    float z  = 1.0f / (1.0f + __expf(-(iz + hz)));
    float nn = tanhf(in_ + r * hn);
    float ho = h[idx];
    h[idx] = nn + z * (ho - nn);
}

// pooled[batch[n], coff + c] += relu(h[n, c])   (float4 granularity)
__global__ void pool_kernel(const float* __restrict__ h, const int* __restrict__ batch,
                            float* __restrict__ pooled, long n, int C4, int totC, int coff) {
    long idx = (long)blockIdx.x * TPB + threadIdx.x;
    if (idx >= n * C4) return;
    long node = idx / C4; int c4 = (int)(idx % C4);
    float4 v = ((const float4*)h)[idx];
    int g = batch[node];
    float* p = pooled + (long)g * totC + coff + c4 * 4;
    unsafeAtomicAdd(p + 0, fmaxf(v.x, 0.0f));
    unsafeAtomicAdd(p + 1, fmaxf(v.y, 0.0f));
    unsafeAtomicAdd(p + 2, fmaxf(v.z, 0.0f));
    unsafeAtomicAdd(p + 3, fmaxf(v.w, 0.0f));
}

__global__ void count_kernel(const int* __restrict__ batch, float* __restrict__ cnt, int n) {
    int i = blockIdx.x * TPB + threadIdx.x;
    if (i < n) unsafeAtomicAdd(&cnt[batch[i]], 1.0f);
}

__global__ void divide_kernel(float* __restrict__ pooled, const float* __restrict__ cnt,
                              int groups, int totC, int C, int coff) {
    long idx = (long)blockIdx.x * TPB + threadIdx.x;
    if (idx >= (long)groups * C) return;
    int g = (int)(idx / C); int c = (int)(idx % C);
    pooled[(long)g * totC + coff + c] /= fmaxf(cnt[g], 1.0f);
}

// BatchNorm over rows: one block per column
__global__ void bn_kernel(float* __restrict__ h, const float* __restrict__ gamma,
                          const float* __restrict__ beta, int rows, int cols) {
    __shared__ float s1[TPB], s2[TPB];
    const int j = blockIdx.x;
    float sum = 0.0f, sq = 0.0f;
    for (int r = threadIdx.x; r < rows; r += TPB) {
        float v = h[(long)r * cols + j];
        sum += v; sq += v * v;
    }
    s1[threadIdx.x] = sum; s2[threadIdx.x] = sq;
    __syncthreads();
    for (int s = TPB / 2; s > 0; s >>= 1) {
        if (threadIdx.x < s) {
            s1[threadIdx.x] += s1[threadIdx.x + s];
            s2[threadIdx.x] += s2[threadIdx.x + s];
        }
        __syncthreads();
    }
    const float mean  = s1[0] / rows;
    const float var   = s2[0] / rows - mean * mean;
    const float scale = rsqrtf(var + 1e-5f) * gamma[j];
    const float shift = beta[j] - mean * scale;
    for (int r = threadIdx.x; r < rows; r += TPB)
        h[(long)r * cols + j] = h[(long)r * cols + j] * scale + shift;
}

// ---------------------------------------------------------------------------
// Host-side orchestration
// ---------------------------------------------------------------------------
static inline long cdivl(long a, long b) { return (a + b - 1) / b; }

static void gemm(hipStream_t s, const float* A, const float* B, const float* bias, float* C,
                 int M, int N, int K, int lda, int ldb, int ldc, int transB, int relu) {
    dim3 g((unsigned)cdivl(N, BN), (unsigned)cdivl(M, BM));
    wmma_gemm_kernel<<<g, TPB, 0, s>>>(A, B, bias, C, M, N, K, lda, ldb, ldc, transB, relu);
}
static void fill(hipStream_t s, float* p, float v, long n) {
    fill_kernel<<<(unsigned)cdivl(n, TPB), TPB, 0, s>>>(p, v, n);
}

extern "C" void kernel_launch(void* const* d_in, const int* in_sizes, int n_in,
                              void* d_out, int out_size, void* d_ws, size_t ws_size,
                              hipStream_t stream)
{
    (void)in_sizes; (void)n_in; (void)out_size; (void)ws_size;
    const int NN = 100000, NE = 200000, NG = 4096, FCD = 2160;

    // jax-pytree (sorted key) flattening of setup_inputs():
    // 0-2 batch1..3 | 3-5 edge_index1..3 | 6 bn_beta | 7 bn_gamma |
    // 8-52 convs c00..c22 x {b_hh,b_ih,w,w_hh,w_ih} |
    // 53 fc1.b 54 fc1.w 55 fc2.b 56 fc2.w 57 fc25.b 58 fc25.w 59 fc3.b 60 fc3.w |
    // 61-63 x1..x3
    const int* batch[3] = {(const int*)d_in[0], (const int*)d_in[1], (const int*)d_in[2]};
    const int* eidx[3]  = {(const int*)d_in[3], (const int*)d_in[4], (const int*)d_in[5]};
    const float* bn_beta  = (const float*)d_in[6];
    const float* bn_gamma = (const float*)d_in[7];
    const float* xin[3] = {(const float*)d_in[61], (const float*)d_in[62], (const float*)d_in[63]};
    auto convp = [&](int comp, int stage, int which) -> const float* {
        return (const float*)d_in[8 + (comp * 3 + stage) * 5 + which];
    }; // which: 0=b_hh 1=b_ih 2=w 3=w_hh 4=w_ih

    // Workspace carve-out (~3.2 GB of the 432 GB device)
    float* wp = (float*)d_ws;
    auto take = [&](size_t nf) { float* q = wp; wp += nf; return q; };
    float* hA     = take((size_t)NN * 720);
    float* hB     = take((size_t)NN * 720);
    float* mb     = take((size_t)NN * 720);
    float* ag     = take((size_t)NN * 720);
    float* gi     = take((size_t)NN * 2160);
    float* gh     = take((size_t)NN * 2160);
    float* pooled = take((size_t)NG * FCD);
    float* cnt    = take((size_t)NG);
    float* f1     = take((size_t)NG * 6480);
    float* f2     = take((size_t)NG * 6480);

    fill(stream, pooled, 0.0f, (long)NG * FCD);

    const int CS[3] = {180, 360, 720};
    for (int comp = 0; comp < 3; ++comp) {
        // pad x[NN,64] -> h[NN,180]
        pad_relu_kernel<<<(unsigned)cdivl((long)NN * 180, TPB), TPB, 0, stream>>>(
            xin[comp], hA, NN, 64, 180, 0);
        float* hcur = hA; float* hoth = hB;
        for (int stage = 0; stage < 3; ++stage) {
            const int C = CS[stage];
            const float* w   = convp(comp, stage, 2);
            const float* whh = convp(comp, stage, 3);
            const float* wih = convp(comp, stage, 4);
            const float* bih = convp(comp, stage, 1);
            const float* bhh = convp(comp, stage, 0);
            for (int l = 0; l < 6; ++l) {
                // m = h @ w[l]
                gemm(stream, hcur, w + (size_t)l * C * C, nullptr, mb,
                     NN, C, C, C, C, C, 0, 0);
                // agg = scatter_add(m[src] -> dst)
                fill(stream, ag, 0.0f, (long)NN * C);
                scatter_add_kernel<<<(unsigned)cdivl((long)NE * (C / 4), TPB), TPB, 0, stream>>>(
                    mb, eidx[comp], ag, NE, C / 4);
                // gi = agg @ w_ih^T + b_ih ; gh = h @ w_hh^T + b_hh
                gemm(stream, ag,   wih, bih, gi, NN, 3 * C, C, C, C, 3 * C, 1, 0);
                gemm(stream, hcur, whh, bhh, gh, NN, 3 * C, C, C, C, 3 * C, 1, 0);
                // h = GRU(gi, gh, h)
                gru_kernel<<<(unsigned)cdivl((long)NN * C, TPB), TPB, 0, stream>>>(
                    gi, gh, hcur, NN, C);
            }
            if (stage < 2) {  // relu + pad to next width, ping-pong buffers
                pad_relu_kernel<<<(unsigned)cdivl((long)NN * CS[stage + 1], TPB), TPB, 0, stream>>>(
                    hcur, hoth, NN, C, CS[stage + 1], 1);
                float* t = hcur; hcur = hoth; hoth = t;
            }
        }
        // relu fused into pooling; mean over graphs
        pool_kernel<<<(unsigned)cdivl((long)NN * (720 / 4), TPB), TPB, 0, stream>>>(
            hcur, batch[comp], pooled, NN, 720 / 4, FCD, comp * 720);
        fill(stream, cnt, 0.0f, NG);
        count_kernel<<<(unsigned)cdivl(NN, TPB), TPB, 0, stream>>>(batch[comp], cnt, NN);
        divide_kernel<<<(unsigned)cdivl((long)NG * 720, TPB), TPB, 0, stream>>>(
            pooled, cnt, NG, FCD, 720, comp * 720);
    }

    // BatchNorm over the 4096-row batch
    bn_kernel<<<FCD, TPB, 0, stream>>>(pooled, bn_gamma, bn_beta, NG, FCD);

    const float* fc1b  = (const float*)d_in[53]; const float* fc1w  = (const float*)d_in[54];
    const float* fc2b  = (const float*)d_in[55]; const float* fc2w  = (const float*)d_in[56];
    const float* fc25b = (const float*)d_in[57]; const float* fc25w = (const float*)d_in[58];
    const float* fc3b  = (const float*)d_in[59]; const float* fc3w  = (const float*)d_in[60];

    gemm(stream, pooled, fc1w,  fc1b,  f1,  NG, 6480, FCD,  FCD,  FCD,  6480, 1, 1);
    gemm(stream, f1,     fc2w,  fc2b,  f2,  NG, 6480, 6480, 6480, 6480, 6480, 1, 1);
    gemm(stream, f2,     fc25w, fc25b, f1,  NG, FCD,  6480, 6480, 6480, FCD,  1, 1);
    gemm(stream, f1,     fc3w,  fc3b,  (float*)d_out, NG, 3, FCD, FCD, FCD, 3, 1, 0);
}
